// ANEAttention_44255343018178
// MI455X (gfx1250) — matrix operand order repaired and checked by reading, not compiled, not run on hardware
//
#include <hip/hip_runtime.h>

typedef __bf16 bf16;
typedef __attribute__((ext_vector_type(16))) __bf16 v16bf;
typedef __attribute__((ext_vector_type(8)))  __bf16 v8bf;
typedef __attribute__((ext_vector_type(8)))  float  v8f;

union ABFrag { v16bf v; v8bf h[2]; };

__device__ __forceinline__ v8f wmma_bf16(v16bf a, v16bf b, v8f c) {
  return __builtin_amdgcn_wmma_f32_16x16x32_bf16(false, a, false, b, (short)0, c,
                                                 false, false);
}

// ---------------------------------------------------------------- fp32 -> bf16
__global__ void cvt_f32_to_bf16(const float* __restrict__ src,
                                bf16* __restrict__ dst, long n) {
  long i = (long)blockIdx.x * blockDim.x + threadIdx.x;
  long stride = (long)gridDim.x * blockDim.x;
  for (; i < n; i += stride) dst[i] = (bf16)src[i];
}

// ------------------------------------------------- fp32 [B,R,C] -> bf16 [B,C,R]
// LDS-tiled 32x32 transpose-cast; coalesced on both global sides.
__global__ __launch_bounds__(256)
void transpose_cast_bf16(const float* __restrict__ src, bf16* __restrict__ dst,
                         int R, int C) {
  __shared__ bf16 tile[32][33];
  const int b = blockIdx.z;
  const int c0 = blockIdx.y * 32;  // row block in src
  const int s0 = blockIdx.x * 32;  // col block in src
  const int tx = threadIdx.x & 31, ty = threadIdx.x >> 5;
  for (int j = ty; j < 32; j += 8)
    tile[j][tx] = (bf16)src[((long)b * R + c0 + j) * C + s0 + tx];
  __syncthreads();
  for (int j = ty; j < 32; j += 8)
    dst[((long)b * C + s0 + j) * R + c0 + tx] = tile[tx][j];
}

// ---------------------------------------------------------------- WMMA GEMM (TN)
// C[M,N](f32 row major) = A[M,K](bf16 row major) * Bt[N,K](bf16 row major)^T
// Both operands are K-contiguous -> every fragment is a direct aligned global
// b128 load; no LDS. Block = 128 threads (4 waves); wave computes 16(M)x64(N),
// sharing one A fragment across 4 WMMAs per k-step.
__global__ __launch_bounds__(128)
void gemm_bf16_wmma_tn(const bf16* __restrict__ A, const bf16* __restrict__ Bt,
                       float* __restrict__ C, int N, int K,
                       long sA, long sB, long sC) {
  const int tid = threadIdx.x;
  const int wid = tid >> 5, lane = tid & 31;
  const int col = lane & 15, hi = lane >> 4;
  const int n0 = blockIdx.x * 64;
  const int m0 = blockIdx.y * 64 + wid * 16;
  const int bz = blockIdx.z;

  const bf16* __restrict__ arow = A + (long)bz * sA + (long)(m0 + col) * K;
  const bf16* __restrict__ brow = Bt + (long)bz * sB + (long)(n0 + col) * K;
  float* __restrict__ Cb = C + (long)bz * sC;

  v8f z = {};
  v8f acc[4] = {z, z, z, z};

  for (int k0 = 0; k0 < K; k0 += 32) {
    if (k0 + 32 < K) {  // speculative prefetch of next k-chunk
      __builtin_prefetch(arow + k0 + 32, 0, 1);
      __builtin_prefetch(brow + k0 + 32, 0, 1);
    }
    ABFrag a;
    a.h[0] = *(const v8bf*)(arow + k0 + 8 * hi);
    a.h[1] = *(const v8bf*)(arow + k0 + 16 + 8 * hi);
#pragma unroll
    for (int j = 0; j < 4; ++j) {
      ABFrag bf;
      bf.h[0] = *(const v8bf*)(brow + (long)j * 16 * K + k0 + 8 * hi);
      bf.h[1] = *(const v8bf*)(brow + (long)j * 16 * K + k0 + 16 + 8 * hi);
      v16bf b = bf.v;
      acc[j] = wmma_bf16(a.v, b, acc[j]);
    }
  }

#pragma unroll
  for (int j = 0; j < 4; ++j)
#pragma unroll
    for (int i = 0; i < 8; ++i)
      Cb[(long)(m0 + i + 8 * hi) * N + n0 + j * 16 + col] = acc[j][i];
}

// ---------------------------------------------------------------- RoPE (+cast)
// x: [B, nh*256, 2048] f32 (channel-major) -> out: [B, nh, 2048, 256] bf16
// (seq-major, d contiguous) so attention fragments are contiguous loads.
__global__ void rope_f32_to_bf16_t(const float* __restrict__ x,
                                   const float* __restrict__ sn,
                                   const float* __restrict__ cs,
                                   bf16* __restrict__ out, int nh, long n) {
  long i0 = (long)blockIdx.x * blockDim.x + threadIdx.x;
  long stride = (long)gridDim.x * blockDim.x;
  for (long i = i0; i < n; i += stride) {
    int s = (int)(i & 2047); long r = i >> 11;
    int d = (int)(r & 127); r >>= 7;
    int h = (int)(r % nh);  int b = (int)(r / nh);
    long xbase = (((long)b * nh + h) * 256 + d) * 2048 + s;
    long tb    = ((long)b * 128 + d) * 2048 + s;
    float x1 = x[xbase], x2 = x[xbase + 128 * 2048];
    float c = cs[tb], sv = sn[tb];
    long obase = (((long)b * nh + h) * 2048 + s) * 256 + d;
    out[obase]       = (bf16)(x1 * c - x2 * sv);
    out[obase + 128] = (bf16)(x2 * c + x1 * sv);
  }
}

// ---------------------------------------------------------------- flash attention
// Qt: [B,8,2048,256] bf16 (roped, d contiguous); Kt: [B,2048,256] bf16 (roped);
// V: [B,256,2048] bf16; mask: [B,2048,2048] f32; O: [B,2048,2048] bf16 laid out
// [b][s][h*256+d] so the final GEMM reads it K-contiguous.
// One wave per 16-query tile; Q fragments preloaded in registers; K/V fragments
// are direct global loads (K+V = 2MB/batch -> L2 resident); only the P matrix
// bounces through 1KB of LDS to reach A-fragment layout.
__global__ __launch_bounds__(32)
void flash_attn_wmma(const bf16* __restrict__ Q, const bf16* __restrict__ Kt,
                     const bf16* __restrict__ V, const float* __restrict__ mask,
                     bf16* __restrict__ O) {
  __shared__ bf16 Pt[16][32];
  const int lane = threadIdx.x;
  const int col = lane & 15, hi = lane >> 4;
  const int q0 = blockIdx.x * 16;
  const int h = blockIdx.y, b = blockIdx.z;
  const long S = 2048, D = 256;

  const bf16* __restrict__ Qb = Q + (((long)b * 8 + h) * S + q0) * D;
  const bf16* __restrict__ Kb = Kt + (long)b * S * D;
  const bf16* __restrict__ Vb = V + (long)b * D * S;
  const float* __restrict__ Mb = mask + (long)b * S * S;
  bf16* __restrict__ Ob = O + ((long)b * S + q0) * 2048 + (long)h * D;

  // preload the 16x256 Q tile as 8 A-fragments (64 VGPRs)
  ABFrag qf[8];
#pragma unroll
  for (int t = 0; t < 8; ++t) {
    qf[t].h[0] = *(const v8bf*)(Qb + (long)col * D + t * 32 + 8 * hi);
    qf[t].h[1] = *(const v8bf*)(Qb + (long)col * D + t * 32 + 16 + 8 * hi);
  }

  float mrun[8], lrun[8];
  v8f zv = {};
  v8f Oacc[16];
#pragma unroll
  for (int i = 0; i < 8; ++i) { mrun[i] = -3.0e38f; lrun[i] = 0.0f; }
#pragma unroll
  for (int nd = 0; nd < 16; ++nd) Oacc[nd] = zv;

  const float scale = 0.0625f;  // 1/sqrt(256)

  for (int k0 = 0; k0 < 2048; k0 += 32) {
    // scores: s[qi][ki], K-fragments straight from global (d contiguous)
    v8f s0 = zv, s1 = zv;
#pragma unroll
    for (int t = 0; t < 8; ++t) {
      ABFrag f0, f1;
      f0.h[0] = *(const v8bf*)(Kb + (long)(k0 + col) * D + t * 32 + 8 * hi);
      f0.h[1] = *(const v8bf*)(Kb + (long)(k0 + col) * D + t * 32 + 16 + 8 * hi);
      f1.h[0] = *(const v8bf*)(Kb + (long)(k0 + 16 + col) * D + t * 32 + 8 * hi);
      f1.h[1] = *(const v8bf*)(Kb + (long)(k0 + 16 + col) * D + t * 32 + 16 + 8 * hi);
      v16bf b0 = f0.v;
      v16bf b1 = f1.v;
      s0 = wmma_bf16(qf[t].v, b0, s0);
      s1 = wmma_bf16(qf[t].v, b1, s1);
    }

    // scale + mask + online softmax (C layout: row = i + 8*hi, col = lane&15)
    float p0[8], p1[8], alpha[8];
#pragma unroll
    for (int i = 0; i < 8; ++i) {
      int m = i + 8 * hi;
      float v0 = s0[i] * scale + Mb[(long)(q0 + m) * S + k0 + col];
      float v1 = s1[i] * scale + Mb[(long)(q0 + m) * S + k0 + 16 + col];
      float rm = fmaxf(v0, v1);
      rm = fmaxf(rm, __shfl_xor(rm, 1, 16));
      rm = fmaxf(rm, __shfl_xor(rm, 2, 16));
      rm = fmaxf(rm, __shfl_xor(rm, 4, 16));
      rm = fmaxf(rm, __shfl_xor(rm, 8, 16));
      float mn = fmaxf(mrun[i], rm);
      float al = __expf(mrun[i] - mn);
      float e0 = __expf(v0 - mn);
      float e1 = __expf(v1 - mn);
      float rs = e0 + e1;
      rs += __shfl_xor(rs, 1, 16);
      rs += __shfl_xor(rs, 2, 16);
      rs += __shfl_xor(rs, 4, 16);
      rs += __shfl_xor(rs, 8, 16);
      lrun[i] = lrun[i] * al + rs;
      mrun[i] = mn;
      alpha[i] = al;
      p0[i] = e0; p1[i] = e1;
    }

#pragma unroll
    for (int nd = 0; nd < 16; ++nd)
#pragma unroll
      for (int i = 0; i < 8; ++i) Oacc[nd][i] *= alpha[i];

    // bounce P through LDS into A-fragment layout
#pragma unroll
    for (int i = 0; i < 8; ++i) {
      Pt[i + 8 * hi][col]      = (bf16)p0[i];
      Pt[i + 8 * hi][16 + col] = (bf16)p1[i];
    }
    __syncthreads();
    ABFrag ap;
    ap.h[0] = *(const v8bf*)(&Pt[col][8 * hi]);
    ap.h[1] = *(const v8bf*)(&Pt[col][16 + 8 * hi]);

    // O[qi][d] += P[qi][ki] * V^T[ki][d]; V fragments contiguous in global
#pragma unroll
    for (int nd = 0; nd < 16; ++nd) {
      ABFrag fv;
      fv.h[0] = *(const v8bf*)(Vb + (long)(nd * 16 + col) * S + k0 + 8 * hi);
      fv.h[1] = *(const v8bf*)(Vb + (long)(nd * 16 + col) * S + k0 + 16 + 8 * hi);
      v16bf vb = fv.v;
      Oacc[nd] = wmma_bf16(ap.v, vb, Oacc[nd]);
    }
    __syncthreads();  // protect Pt before next iteration's writes
  }

#pragma unroll
  for (int i = 0; i < 8; ++i) lrun[i] = 1.0f / lrun[i];
#pragma unroll
  for (int nd = 0; nd < 16; ++nd)
#pragma unroll
    for (int i = 0; i < 8; ++i)
      Ob[(long)(i + 8 * hi) * 2048 + nd * 16 + col] =
          (bf16)(Oacc[nd][i] * lrun[i]);
}

// ----------------------------------------------------------------------------
extern "C" void kernel_launch(void* const* d_in, const int* in_sizes, int n_in,
                              void* d_out, int out_size, void* d_ws, size_t ws_size,
                              hipStream_t stream) {
  (void)in_sizes; (void)n_in; (void)out_size; (void)ws_size;
  const float* Xq   = (const float*)d_in[0];
  const float* Xkv  = (const float*)d_in[1];
  const float* sinq = (const float*)d_in[2];
  const float* cosq = (const float*)d_in[3];
  const float* sink = (const float*)d_in[4];
  const float* cosk = (const float*)d_in[5];
  const float* mask = (const float*)d_in[6];
  const float* Wq   = (const float*)d_in[7];
  const float* Wk   = (const float*)d_in[8];
  const float* Wv   = (const float*)d_in[9];
  const float* Wo   = (const float*)d_in[10];
  float* out = (float*)d_out;

  const long B = 2, S = 2048, HID = 2048, H = 8, D = 256, HD = 2048;

  char* ws = (char*)d_ws;
  size_t off = 0;
  auto alloc = [&](size_t bytes) {
    void* p = ws + off;
    off += (bytes + 255) & ~(size_t)255;
    return p;
  };
  bf16* xq_t   = (bf16*)alloc(B * S * HID * 2);   // [B,S,HID]
  bf16* xkv_t  = (bf16*)alloc(B * S * HID * 2);   // [B,S,HID]
  bf16* wq_bf  = (bf16*)alloc(HD * HID * 2);
  bf16* wk_bf  = (bf16*)alloc(D * HID * 2);
  bf16* wv_bf  = (bf16*)alloc(D * HID * 2);
  bf16* wo_bf  = (bf16*)alloc(HID * HD * 2);
  float* q_f32 = (float*)alloc(B * HD * S * 4);   // [B,HD,S]
  float* k_f32 = (float*)alloc(B * D * S * 4);    // [B,D,S]
  float* v_f32 = (float*)alloc(B * D * S * 4);
  bf16* q_t    = (bf16*)alloc(B * HD * S * 2);    // [B,H,S,256] roped
  bf16* k_t    = (bf16*)alloc(B * D * S * 2);     // [B,S,256] roped
  bf16* v_bf   = (bf16*)alloc(B * D * S * 2);     // [B,256,S]
  bf16* o_t    = (bf16*)alloc(B * S * HD * 2);    // [B,S,HD]

  // 1) layout prep: transpose-cast activations, cast weights
  transpose_cast_bf16<<<dim3(S / 32, HID / 32, B), 256, 0, stream>>>(
      Xq, xq_t, (int)HID, (int)S);
  transpose_cast_bf16<<<dim3(S / 32, HID / 32, B), 256, 0, stream>>>(
      Xkv, xkv_t, (int)HID, (int)S);
  cvt_f32_to_bf16<<<1024, 256, 0, stream>>>(Wq, wq_bf, HD * HID);
  cvt_f32_to_bf16<<< 256, 256, 0, stream>>>(Wk, wk_bf, D * HID);
  cvt_f32_to_bf16<<< 256, 256, 0, stream>>>(Wv, wv_bf, D * HID);
  cvt_f32_to_bf16<<<1024, 256, 0, stream>>>(Wo, wo_bf, HID * HD);

  // 2) projections (LDS-free WMMA GEMMs)
  gemm_bf16_wmma_tn<<<dim3(S / 64, HD / 64, B), 128, 0, stream>>>(
      wq_bf, xq_t, q_f32, (int)S, (int)HID, 0L, S * HID, HD * S);
  gemm_bf16_wmma_tn<<<dim3(S / 64, D / 64, B), 128, 0, stream>>>(
      wk_bf, xkv_t, k_f32, (int)S, (int)HID, 0L, S * HID, D * S);
  gemm_bf16_wmma_tn<<<dim3(S / 64, D / 64, B), 128, 0, stream>>>(
      wv_bf, xkv_t, v_f32, (int)S, (int)HID, 0L, S * HID, D * S);

  // 3) RoPE + cast, writing d-contiguous layouts for attention fragments
  rope_f32_to_bf16_t<<<2048, 256, 0, stream>>>(q_f32, sinq, cosq, q_t, (int)H,
                                               B * H * 128 * S);
  rope_f32_to_bf16_t<<< 512, 256, 0, stream>>>(k_f32, sink, cosk, k_t, 1,
                                               B * 1 * 128 * S);
  cvt_f32_to_bf16<<<1024, 256, 0, stream>>>(v_f32, v_bf, B * D * S);

  // 4) fused flash attention (WMMA + online softmax)
  flash_attn_wmma<<<dim3(S / 16, H, B), 32, 0, stream>>>(q_t, k_t, v_bf, mask,
                                                         o_t);

  // 5) output projection (reads o_t K-contiguous)
  gemm_bf16_wmma_tn<<<dim3(S / 64, HID / 64, B), 128, 0, stream>>>(
      wo_bf, o_t, out, (int)S, (int)HD, 0L, S * HD, HID * S);
}
